// SimpleMetaNet_11768210391588
// MI455X (gfx1250) — matrix operand order repaired and checked by reading, not compiled, not run on hardware
//
#include <hip/hip_runtime.h>
#include <math.h>

#define Bn    64
#define Mn    2048
#define Dn    256
#define Hn    512
#define Kn    64
#define XLD   (Dn + 1)       // 257 floats per token row in x
#define MINV  (1.0f / (float)Mn)

typedef float     v8f  __attribute__((ext_vector_type(8)));
typedef float     v2f  __attribute__((ext_vector_type(2)));
typedef _Float16  v16h __attribute__((ext_vector_type(16)));

// ---------------------------------------------------------------------------
// WMMA helpers
// ---------------------------------------------------------------------------
__device__ __forceinline__ v8f wmma_f32(v2f a, v2f b, v8f c) {
    return __builtin_amdgcn_wmma_f32_16x16x4_f32(
        false, a, false, b, (short)0, c, false, false);
}
__device__ __forceinline__ v8f wmma_f16(v16h a, v16h b, v8f c) {
    return __builtin_amdgcn_wmma_f32_16x16x32_f16(
        false, a, false, b, (short)0, c, false, false);
}

// ---------------------------------------------------------------------------
// Convert + transpose k1_w0 (256x512 f32, row-major) -> w0t (512x256 f16,
// col-major) so B fragments become contiguous 32B loads.
// ---------------------------------------------------------------------------
__global__ __launch_bounds__(256) void w0t_kernel(const float* __restrict__ w0,
                                                  _Float16* __restrict__ w0t) {
    int c = blockIdx.x;     // 0..511
    int k = threadIdx.x;    // 0..255
    w0t[(size_t)c * Dn + k] = (_Float16)w0[(size_t)k * Hn + c];
}

// ---------------------------------------------------------------------------
// sum of labels per batch: sumlab[b] = sum_m x[b,m,256]
// ---------------------------------------------------------------------------
__global__ __launch_bounds__(256) void sumlab_kernel(const float* __restrict__ x,
                                                     float* __restrict__ sumlab) {
    int b = blockIdx.x, tid = threadIdx.x;
    __shared__ float red[256];
    float s = 0.f;
    for (int m = tid; m < Mn; m += 256)
        s += x[((size_t)(b * Mn + m)) * XLD + Dn];
    red[tid] = s; __syncthreads();
    for (int st = 128; st > 0; st >>= 1) {
        if (tid < st) red[tid] += red[tid + st];
        __syncthreads();
    }
    if (tid == 0) sumlab[b] = red[0];
}

// ---------------------------------------------------------------------------
// Stage 1 (dominant, f16 WMMA): partial s1[b][rc][c] over 128 rows of
// label * relu(f@W0+b0). grid = B*16*8, block = 256 (8 waves).
// Wave = one 16-row tile; A fragments (full K=256) preloaded into registers
// and reused across 4 col-tiles; B from f16 col-major w0t (32B frags).
//
// Layouts used here (16-bit, wave32, same k order in A and B): A v16h lane L: K = A0+{0..15},
// A0=(L>>4)*16, M=L&15. B v16h lane L: K = Bsel+{0..15}, Bsel=(L>>4)*16,
// N=L&15. C/D v8f: VGPR i -> (M = i + (L>>4)*8, N = L&15).
// ---------------------------------------------------------------------------
__global__ __launch_bounds__(256) void k1_stage1_f16_kernel(
    const float* __restrict__ x, const _Float16* __restrict__ w0t,
    const float* __restrict__ b0, float* __restrict__ part_out /*[B][16][512]*/) {

    int blk = blockIdx.x;
    int b  = blk >> 7;
    int rc = (blk >> 3) & 15;
    int cc = blk & 7;
    int tid = threadIdx.x;
    int wave = tid >> 5;
    int lane = tid & 31;

    __shared__ float part[8][64];

    int rowbase = rc * 128 + wave * 16;
    int cbase   = cc * 64;
    int rl      = lane & 15;
    int hi      = lane >> 4;     // lane half
    int A0      = hi * 16;       // A K sub-offset
    int Bsel    = hi * 16;       // B K sub-offset

    const float* arow = x + ((size_t)(b * Mn + rowbase + rl)) * XLD;

    // Preload all A fragments for K=256 (8 steps of 32), f32 -> f16 on the fly
    v16h af[8];
    #pragma unroll
    for (int s = 0; s < 8; ++s) {
        int kb = s * 32;
        #pragma unroll
        for (int j = 0; j < 8; ++j) {
            af[s][j]     = (_Float16)arow[kb + A0 + j];
            af[s][8 + j] = (_Float16)arow[kb + A0 + 8 + j];
        }
    }
    // labels for this lane's 8 epilogue rows
    float labv[8];
    #pragma unroll
    for (int i = 0; i < 8; ++i)
        labv[i] = x[((size_t)(b * Mn + rowbase + hi * 8 + i)) * XLD + Dn];

    for (int ct = 0; ct < 4; ++ct) {
        int c0 = cbase + ct * 16;
        const _Float16* wcol = w0t + (size_t)(c0 + rl) * Dn + Bsel;
        v8f acc = {};
        #pragma unroll
        for (int s = 0; s < 8; ++s) {
            v16h bf = *(const v16h*)(wcol + s * 32);   // 32B aligned
            acc = wmma_f16(af[s], bf, acc);
        }
        // epilogue: relu(+bias), weight by label, fold 16 rows -> per-col
        float bcol = b0[c0 + rl];
        float p = 0.f;
        #pragma unroll
        for (int i = 0; i < 8; ++i) {
            float t = acc[i] + bcol;
            t = t > 0.f ? t : 0.f;
            p += t * labv[i];
        }
        p += __shfl_xor(p, 16);
        if (lane < 16) part[wave][ct * 16 + rl] = p;
    }
    __syncthreads();
    if (tid < 64) {
        float s = 0.f;
        #pragma unroll
        for (int w = 0; w < 8; ++w) s += part[w][tid];
        part_out[(((size_t)b * 16 + rc)) * Hn + cbase + tid] = s;
    }
}

// s[b][c] = sum over 16 rowchunks of partials (fixed order => deterministic)
__global__ __launch_bounds__(256) void reduce_s1_kernel(const float* __restrict__ part,
                                                        float* __restrict__ s) {
    int i = blockIdx.x * 256 + threadIdx.x;
    if (i >= Bn * Hn) return;
    int b = i >> 9, c = i & 511;
    float v = 0.f;
    for (int rc = 0; rc < 16; ++rc)
        v += part[(((size_t)b * 16 + rc)) * Hn + c];
    s[i] = v;
}

// ---------------------------------------------------------------------------
// Generic 64-row f32 WMMA GEMM with flexible epilogue:
//   C[r][c] = act( alpha*(A@W)[r][c] + bias[c]*(rowscale? rowscale[r]*rs_mul : 1) )
//             + beta*D[r][c]
// ---------------------------------------------------------------------------
__global__ __launch_bounds__(128) void gemm64_kernel(
    const float* __restrict__ A, int lda, int K,
    const float* __restrict__ W, int N,
    const float* __restrict__ bias,
    const float* __restrict__ rowscale, float rs_mul,
    const float* __restrict__ Dmat, int ldd, float beta,
    float alpha, int do_relu,
    float* __restrict__ C, int ldc) {

    int ntc  = N >> 4;
    int tile = blockIdx.x * 4 + (threadIdx.x >> 5);
    if (tile >= 4 * ntc) return;                   // uniform per wave
    int rt = tile / ntc;
    int ct = tile % ntc;
    int lane = threadIdx.x & 31;
    int rl = lane & 15;
    int kpair = (lane >> 4) << 1;
    int rowHalf = lane >> 4;
    int r0 = rt * 16, c0 = ct * 16;

    v8f acc = {};
    const float* arow = A + (size_t)(r0 + rl) * lda;
    const float* wcol = W + c0 + rl;
    for (int k = 0; k < K; k += 4) {
        v2f a, bb;
        a.x  = arow[k + kpair];
        a.y  = arow[k + kpair + 1];
        bb.x = wcol[(size_t)(k + kpair) * N];
        bb.y = wcol[(size_t)(k + kpair + 1) * N];
        acc = wmma_f32(a, bb, acc);
    }
    int col = c0 + rl;
    float bcol = bias ? bias[col] : 0.f;
    #pragma unroll
    for (int i = 0; i < 8; ++i) {
        int row = r0 + rowHalf * 8 + i;
        float rs = rowscale ? rowscale[row] * rs_mul : 1.f;
        float v = acc[i] * alpha + bcol * rs;
        if (do_relu) v = v > 0.f ? v : 0.f;
        if (Dmat) v += beta * Dmat[(size_t)row * ldd + col];
        C[(size_t)row * ldc + col] = v;
    }
}

// ---------------------------------------------------------------------------
// Stage 3 (per batch): gumbel + softmax over M, top-64 (lowest-index ties),
// mask values ((1-o)+o), selected-label sum, msg copy into z[:,512:].
// ---------------------------------------------------------------------------
__global__ __launch_bounds__(256) void stage3_kernel(
    const float* __restrict__ lin1 /*[B][2560]*/, const float* __restrict__ u,
    const float* __restrict__ x,
    int* __restrict__ idx_out, float* __restrict__ maskval_out,
    float* __restrict__ sel_lab, float* __restrict__ z /*[B][1024]*/) {

    __shared__ float odds[Mn];
    __shared__ float work[Mn];
    __shared__ float redv[256];
    __shared__ int   redi[256];

    int b = blockIdx.x, tid = threadIdx.x;
    const float* lrow = lin1 + (size_t)b * (Hn + Mn);

    float lmax = -__builtin_inff();
    for (int m = tid; m < Mn; m += 256) {
        float g = -logf(-logf(u[(size_t)b * Mn + m]));
        float l = lrow[m] + g;                     // TAU == 1
        odds[m] = l;
        lmax = l > lmax ? l : lmax;
    }
    redv[tid] = lmax; __syncthreads();
    for (int st = 128; st > 0; st >>= 1) {
        if (tid < st) redv[tid] = redv[tid] > redv[tid + st] ? redv[tid] : redv[tid + st];
        __syncthreads();
    }
    float gmax = redv[0]; __syncthreads();

    float lsum = 0.f;
    for (int m = tid; m < Mn; m += 256) {
        float e = expf(odds[m] - gmax);
        odds[m] = e;
        lsum += e;
    }
    redv[tid] = lsum; __syncthreads();
    for (int st = 128; st > 0; st >>= 1) {
        if (tid < st) redv[tid] += redv[tid + st];
        __syncthreads();
    }
    float inv = 1.f / redv[0]; __syncthreads();
    for (int m = tid; m < Mn; m += 256) {
        float o = odds[m] * inv;
        odds[m] = o;
        work[m] = o;
    }
    __syncthreads();

    for (int t = 0; t < Kn; ++t) {
        float bv = -__builtin_inff(); int bi = Mn;
        for (int m = tid; m < Mn; m += 256) {
            float v = work[m];
            if (v > bv || (v == bv && m < bi)) { bv = v; bi = m; }
        }
        redv[tid] = bv; redi[tid] = bi; __syncthreads();
        for (int st = 128; st > 0; st >>= 1) {
            if (tid < st) {
                if (redv[tid + st] > redv[tid] ||
                    (redv[tid + st] == redv[tid] && redi[tid + st] < redi[tid])) {
                    redv[tid] = redv[tid + st]; redi[tid] = redi[tid + st];
                }
            }
            __syncthreads();
        }
        int sel = redi[0];
        if (tid == 0) {
            work[sel] = -__builtin_inff();
            float o = odds[sel];
            idx_out[b * Kn + t] = sel;
            maskval_out[b * Kn + t] = (1.0f - o) + o;
        }
        __syncthreads();
    }

    if (tid < Kn) {
        int m = idx_out[b * Kn + tid];
        redv[tid] = x[((size_t)(b * Mn + m)) * XLD + Dn];
    }
    __syncthreads();
    if (tid < 32) redv[tid] += redv[tid + 32];
    __syncthreads();
    if (tid < 16) redv[tid] += redv[tid + 16];
    __syncthreads();
    if (tid < 8) redv[tid] += redv[tid + 8];
    __syncthreads();
    if (tid < 4) redv[tid] += redv[tid + 4];
    __syncthreads();
    if (tid == 0) sel_lab[b] = redv[0] + redv[1] + redv[2] + redv[3];

    for (int c = tid; c < Hn; c += 256)
        z[(size_t)b * (2 * Hn) + Hn + c] = lrow[Mn + c];
}

// ---------------------------------------------------------------------------
// Stage 4 (per batch): gathered 64-row GEMM through k2_w0 (f32 WMMA for
// precision; ~1 GFLOP total). Adds constant relu(b0) term for unselected rows.
// ---------------------------------------------------------------------------
__global__ __launch_bounds__(256) void k2_stage4_kernel(
    const float* __restrict__ x, const float* __restrict__ w0,
    const float* __restrict__ b0, const int* __restrict__ idx,
    const float* __restrict__ maskval, const float* __restrict__ sel_lab,
    const float* __restrict__ sumlab, float* __restrict__ s2_total) {

    int b = blockIdx.x, tid = threadIdx.x;
    int wave = tid >> 5, lane = tid & 31;
    __shared__ float part[4][Hn];

    int rt = wave & 3;
    int cgrp = wave >> 2;
    int rl = lane & 15;
    int kpair = (lane >> 4) << 1;
    int rowHalf = lane >> 4;

    int j = rt * 16 + rl;
    int m = idx[b * Kn + j];
    const float* arow = x + ((size_t)(b * Mn + m)) * XLD;

    for (int ct = cgrp * 16; ct < cgrp * 16 + 16; ++ct) {
        int c0 = ct * 16;
        v8f acc = {};
        const float* wcol = w0 + c0 + rl;
        for (int k = 0; k < Dn; k += 4) {
            v2f a, bb;
            a.x  = arow[k + kpair];
            a.y  = arow[k + kpair + 1];
            bb.x = wcol[(size_t)(k + kpair) * Hn];
            bb.y = wcol[(size_t)(k + kpair + 1) * Hn];
            acc = wmma_f32(a, bb, acc);
        }
        float bcol = b0[c0 + rl];
        float p = 0.f;
        #pragma unroll
        for (int i = 0; i < 8; ++i) {
            int jj = rt * 16 + rowHalf * 8 + i;
            int mm = idx[b * Kn + jj];
            float mv = maskval[b * Kn + jj];
            float lab = x[((size_t)(b * Mn + mm)) * XLD + Dn];
            float t = mv * acc[i] + bcol;
            t = t > 0.f ? t : 0.f;
            p += t * lab;
        }
        p += __shfl_xor(p, 16);
        if (lane < 16) part[rt][c0 + rl] = p;
    }
    __syncthreads();
    float rest = sumlab[b] - sel_lab[b];
    for (int c = tid; c < Hn; c += 256) {
        float s = part[0][c] + part[1][c] + part[2][c] + part[3][c];
        float r0 = b0[c];
        r0 = r0 > 0.f ? r0 : 0.f;
        s2_total[(size_t)b * Hn + c] = s + rest * r0;
    }
}

// ---------------------------------------------------------------------------
// Host-side orchestration
// ---------------------------------------------------------------------------
extern "C" void kernel_launch(void* const* d_in, const int* in_sizes, int n_in,
                              void* d_out, int out_size, void* d_ws, size_t ws_size,
                              hipStream_t stream) {
    const float* x     = (const float*)d_in[0];
    const float* u     = (const float*)d_in[1];
    const float* k1_w0 = (const float*)d_in[2];
    const float* k1_b0 = (const float*)d_in[3];
    const float* k1_w1 = (const float*)d_in[4];
    const float* k1_b1 = (const float*)d_in[5];
    const float* k2_w0 = (const float*)d_in[6];
    const float* k2_b0 = (const float*)d_in[7];
    const float* k2_w1 = (const float*)d_in[8];
    const float* k2_b1 = (const float*)d_in[9];
    const float* m1_w0 = (const float*)d_in[10];
    const float* m1_b0 = (const float*)d_in[11];
    const float* m1_w1 = (const float*)d_in[12];
    const float* m1_b1 = (const float*)d_in[13];
    const float* m2_w0 = (const float*)d_in[14];
    const float* m2_b0 = (const float*)d_in[15];
    const float* m2_w1 = (const float*)d_in[16];
    const float* m2_b1 = (const float*)d_in[17];
    const float* l1_w  = (const float*)d_in[18];
    const float* l1_b  = (const float*)d_in[19];
    const float* l2_w  = (const float*)d_in[20];
    const float* l2_b  = (const float*)d_in[21];

    float* f = (float*)d_ws;
    _Float16* w0t = (_Float16*)f;  f += (Hn * Dn) / 2;   // 512x256 f16
    float* sumlab = f;  f += Bn;
    float* s1part = f;  f += (size_t)Bn * 16 * Hn;
    float* sbuf   = f;  f += Bn * Hn;
    float* xt     = f;  f += Bn * Hn;
    float* t1     = f;  f += Bn * Hn;
    float* xtm    = f;  f += Bn * Hn;
    float* lin1   = f;  f += (size_t)Bn * (Hn + Mn);
    float* maskv  = f;  f += Bn * Kn;
    float* sellab = f;  f += Bn;
    float* z      = f;  f += Bn * 2 * Hn;
    float* s2t    = f;  f += Bn * Hn;
    float* t2     = f;  f += Bn * Hn;
    float* yz     = f;  f += Bn * 2 * Hn;
    int*   idx    = (int*)f;

    // 0) weight prep: k1_w0 -> f16 col-major
    w0t_kernel<<<Hn, 256, 0, stream>>>(k1_w0, w0t);

    // 1) label sums
    sumlab_kernel<<<Bn, 256, 0, stream>>>(x, sumlab);

    // 2) heavy fused GEMM + weighted reduce for k1 path (f16 WMMA, f32 acc)
    k1_stage1_f16_kernel<<<Bn * 16 * 8, 256, 0, stream>>>(x, w0t, k1_b0, s1part);
    reduce_s1_kernel<<<(Bn * Hn + 255) / 256, 256, 0, stream>>>(s1part, sbuf);

    // 3) xt = (s/M) @ k1_w1 + (sumlab/M)*k1_b1
    gemm64_kernel<<<32, 128, 0, stream>>>(sbuf, Hn, Hn, k1_w1, Hn, k1_b1,
                                          sumlab, MINV, nullptr, 0, 0.f,
                                          MINV, 0, xt, Hn);
    // 4) t1 = relu(xt @ m1_w0 + m1_b0)
    gemm64_kernel<<<32, 128, 0, stream>>>(xt, Hn, Hn, m1_w0, Hn, m1_b0,
                                          nullptr, 0.f, nullptr, 0, 0.f,
                                          1.f, 1, t1, Hn);
    // 5) xtm = t1 @ m1_w1 + m1_b1 + xt(skip)
    gemm64_kernel<<<32, 128, 0, stream>>>(t1, Hn, Hn, m1_w1, Hn, m1_b1,
                                          nullptr, 0.f, xt, Hn, 1.f,
                                          1.f, 0, xtm, Hn);
    // 6) lin1 = xtm @ l1_w + l1_b   (64 x 2560)
    gemm64_kernel<<<160, 128, 0, stream>>>(xtm, Hn, Hn, l1_w, Hn + Mn, l1_b,
                                           nullptr, 0.f, nullptr, 0, 0.f,
                                           1.f, 0, lin1, Hn + Mn);
    // 7) gumbel softmax + top-64 + mask + msg copy
    stage3_kernel<<<Bn, 256, 0, stream>>>(lin1, u, x, idx, maskv, sellab, z);

    // 8) sparse k2 path: gathered GEMM + weighted reduce + constant term
    k2_stage4_kernel<<<Bn, 256, 0, stream>>>(x, k2_w0, k2_b0, idx, maskv,
                                             sellab, sumlab, s2t);
    // 9) xt2 = (s2/M) @ k2_w1 + (sumlab/M)*k2_b1  -> z[:, :512]
    gemm64_kernel<<<32, 128, 0, stream>>>(s2t, Hn, Hn, k2_w1, Hn, k2_b1,
                                          sumlab, MINV, nullptr, 0, 0.f,
                                          MINV, 0, z, 2 * Hn);
    // 10) t2 = relu(z @ m2_w0 + m2_b0)   (K = 1024)
    gemm64_kernel<<<32, 128, 0, stream>>>(z, 2 * Hn, 2 * Hn, m2_w0, Hn, m2_b0,
                                          nullptr, 0.f, nullptr, 0, 0.f,
                                          1.f, 1, t2, Hn);
    // 11) yz = relu(t2 @ m2_w1 + m2_b1) + z   (64 x 1024)
    gemm64_kernel<<<64, 128, 0, stream>>>(t2, Hn, Hn, m2_w1, 2 * Hn, m2_b1,
                                          nullptr, 0.f, z, 2 * Hn, 1.f,
                                          1.f, 1, yz, 2 * Hn);
    // 12) out = yz @ l2_w + l2_b   (64 x 512)
    gemm64_kernel<<<32, 128, 0, stream>>>(yz, 2 * Hn, 2 * Hn, l2_w, Hn, l2_b,
                                          nullptr, 0.f, nullptr, 0, 0.f,
                                          1.f, 0, (float*)d_out, Hn);
}
